// ACE_Descriptor_81913616269878
// MI455X (gfx1250) — matrix operand order repaired and checked by reading, not compiled, not run on hardware
//
#include <hip/hip_runtime.h>
#include <hip/hip_bf16.h>

// CDNA5 / gfx1250 fused ACE descriptor.
// Edge kernel: 16 edges per wave; all GEMMs via v_wmma_f32_16x16x32_bf16.
// Weights staged in LDS pre-swizzled to B-fragment order -> 2x ds_load_b128
// per fragment. A-fragments read as 2x b128 from row-major staging. Scatter
// via global_atomic_add_f32 (target is L2-resident: 64 MB < 192 MB L2).
// Node kernel: 16 nodes per wave, B0 @ Wmix0 via WMMA, in-place add.

typedef __attribute__((ext_vector_type(16))) __bf16        v16bf;
typedef __attribute__((ext_vector_type(8)))  float          v8f;
typedef __attribute__((ext_vector_type(8)))  unsigned short us8;

union AFrag {
  v16bf v;
  us8   h[2];
  unsigned short u[16];
};

__device__ __forceinline__ unsigned short f32_to_bf16(float f) {
  unsigned int b = __float_as_uint(f);
  b += 0x7FFFu + ((b >> 16) & 1u);       // round-to-nearest-even
  return (unsigned short)(b >> 16);
}

// Destination index (in ushorts) for element (k,n) of a K x (16*T) weight
// matrix stored in B-fragment order: fragment (s,t), lane = (kk>>4)<<4 | (n&15),
// element i = kk&15, all 16 elements of a lane contiguous.
__device__ __forceinline__ int bfrag_idx(int k, int n, int T) {
  int s  = k >> 5;
  int kk = k & 31;
  int lane = ((kk >> 4) << 4) | (n & 15);
  int t  = n >> 4;
  return ((((s * T + t) << 5) + lane) << 4) | (kk & 15);
}

// Load one B fragment (2 x b128) for (s-th K-step, t-th N-tile).
__device__ __forceinline__ void load_bfrag(AFrag& b, const unsigned short* buf,
                                           int s, int t, int T, int lane) {
  const us8* p = (const us8*)(buf + ((((s * T + t) << 5) + lane) << 4));
  b.h[0] = p[0];
  b.h[1] = p[1];
}

// Load one A fragment (2 x b128) from a row-major (16 x rowStride) bf16 buffer.
__device__ __forceinline__ void load_afrag(AFrag& a, const unsigned short* buf,
                                           int row, int rowStride, int K0, int hi) {
  a.h[0] = *(const us8*)(buf + row * rowStride + K0 + 16 * hi);
  a.h[1] = *(const us8*)(buf + row * rowStride + K0 + 16 * hi + 8);
}

#define WAVES 8
#define TPB   (WAVES * 32)

// ---------------------------------------------------------------------------
// Kernel 1: per-edge fused pipeline + atomic scatter into out (N,320)
// ---------------------------------------------------------------------------
__global__ __launch_bounds__(TPB)
void ace_edge_kernel(const float* __restrict__ node_attrs,
                     const int*   __restrict__ send,
                     const int*   __restrict__ recv,
                     const float* __restrict__ edge_vec,
                     const float* __restrict__ edge_len,
                     const float* __restrict__ W1,
                     const float* __restrict__ W2,
                     const float* __restrict__ Wtp0,
                     const float* __restrict__ Wtp1,
                     float* __restrict__ out,
                     int E)
{
  __shared__ unsigned short sW1[32 * 64];         //  4 KB (K padded 8->32, frag order)
  __shared__ unsigned short sW2[64 * 128];        // 16 KB (frag order)
  __shared__ unsigned short sT0[128 * 128];       // 32 KB (frag order)
  __shared__ unsigned short sT1[128 * 64];        // 16 KB (frag order)
  __shared__ unsigned short sSC[WAVES][16 * 128]; // 32 KB per-wave row-major staging
  __shared__ int   sSend[WAVES][16];
  __shared__ int   sRecv[WAVES][16];
  __shared__ float sY1[WAVES][16][3];

  const int tid  = threadIdx.x;
  const int wv   = tid >> 5;
  const int lane = tid & 31;
  const int hi   = lane >> 4;   // wave half
  const int cl   = lane & 15;   // A: row M, B/D: column N

  // Cooperative f32 -> bf16 weight staging, swizzled to B-fragment order.
  for (int idx = tid; idx < 32*64; idx += TPB) {
    int k = idx >> 6, n = idx & 63;
    float w = (k < 8) ? W1[k * 64 + n] : 0.f;    // zero-pad K: kills divergent loads
    sW1[bfrag_idx(k, n, 4)] = f32_to_bf16(w);
  }
  for (int idx = tid; idx < 64*128; idx += TPB) {
    int k = idx >> 7, n = idx & 127;
    sW2[bfrag_idx(k, n, 8)] = f32_to_bf16(W2[idx]);
  }
  for (int idx = tid; idx < 128*128; idx += TPB) {
    int k = idx >> 7, n = idx & 127;
    sT0[bfrag_idx(k, n, 8)] = f32_to_bf16(Wtp0[idx]);
  }
  for (int idx = tid; idx < 128*64; idx += TPB) {
    int k = idx >> 6, n = idx & 63;
    sT1[bfrag_idx(k, n, 4)] = f32_to_bf16(Wtp1[idx]);
  }

  const int e0 = blockIdx.x * (WAVES * 16) + wv * 16;

  // --- Radial basis directly in A-fragment layout (16x32, K=8 valid) ------
  AFrag aRad;
  #pragma unroll
  for (int i = 0; i < 16; i++) aRad.u[i] = 0;

  if (lane < 16) {
    const int e = e0 + lane;
    int   sn = 0, rc = 0;
    float len = 0.f, vx = 0.f, vy = 0.f, vz = 0.f;
    if (e < E) {
      sn  = send[e];
      rc  = recv[e];
      len = edge_len[e];
      vx  = edge_vec[3*e+0];
      vy  = edge_vec[3*e+1];
      vz  = edge_vec[3*e+2];
      __builtin_prefetch(&node_attrs[(size_t)sn * 128], 0, 3); // global_prefetch
    }
    sSend[wv][lane] = sn;
    sRecv[wv][lane] = rc;
    float n2 = vx*vx + vy*vy + vz*vz;
    float rn = (n2 > 0.f) ? (1.7320508f * __builtin_amdgcn_rsqf(n2)) : 0.f;
    sY1[wv][lane][0] = vy * rn;                  // Y1 = sqrt3 * unit[[1,2,0]]
    sY1[wv][lane][1] = vz * rn;
    sY1[wv][lane][2] = vx * rn;
    if (e < E && len > 0.f && len < 5.0f) {
      float c = 0.22360680f / len;               // sqrt(2/R_MAX)/sqrt(8)/x
      #pragma unroll
      for (int i = 0; i < 8; i++)
        aRad.u[i] = f32_to_bf16(c * __sinf((float)(i+1) * 0.62831853f * len));
    }
  }
  __syncthreads();

  unsigned short* sc = sSC[wv];                  // 16 x 128 bf16 staging

  // Hoist per-edge metadata into registers (rows handled by this lane).
  int   snReg[8], obReg[8];
  float y1Reg[8][3];
  bool  okReg[8];
  #pragma unroll
  for (int r = 0; r < 8; r++) {
    int m = r + 8*hi;
    snReg[r] = sSend[wv][m];
    obReg[r] = sRecv[wv][m] * 320;
    y1Reg[r][0] = sY1[wv][m][0];
    y1Reg[r][1] = sY1[wv][m][1];
    y1Reg[r][2] = sY1[wv][m][2];
    okReg[r] = (e0 + m) < E;
  }

  // --- Phase 1: h = 1.679177 * silu(rad @ W1/sqrt8), staged to LDS --------
  #pragma unroll
  for (int t = 0; t < 4; t++) {
    AFrag b;
    load_bfrag(b, sW1, 0, t, 4, lane);
    v8f acc = {};
    acc = __builtin_amdgcn_wmma_f32_16x16x32_bf16(false, aRad.v, false, b.v,
                                                  (short)0, acc, false, false);
    #pragma unroll
    for (int r = 0; r < 8; r++) {
      float y   = acc[r];
      float sig = __builtin_amdgcn_rcpf(1.0f + __expf(-y));   // fast sigmoid
      sc[(r + 8*hi)*128 + 16*t + cl] = f32_to_bf16(1.679177f * y * sig);
    }
  }

  // --- Phase 2: R_n = h @ W2/8 ; node_feats = gather(sender)*R_n ----------
  AFrag aH[2];
  #pragma unroll
  for (int s = 0; s < 2; s++) load_afrag(aH[s], sc, cl, 128, 32*s, hi);

  #pragma unroll
  for (int t = 0; t < 8; t++) {
    v8f acc = {};
    #pragma unroll
    for (int s = 0; s < 2; s++) {
      AFrag b;
      load_bfrag(b, sW2, s, t, 8, lane);
      acc = __builtin_amdgcn_wmma_f32_16x16x32_bf16(false, aH[s].v, false, b.v,
                                                    (short)0, acc, false, false);
    }
    #pragma unroll
    for (int r = 0; r < 8; r++) {                // coalesced 16-float row reads
      float na = node_attrs[(size_t)snReg[r]*128 + 16*t + cl];
      sc[(r + 8*hi)*128 + 16*t + cl] = f32_to_bf16(na * acc[r] * 0.125f);
    }
  }

  // --- Phase 3: f0 = nf @ Wtp0 * inv -> atomic scatter --------------------
  AFrag aN[4];
  #pragma unroll
  for (int s = 0; s < 4; s++) load_afrag(aN[s], sc, cl, 128, 32*s, hi);

  const float inv = 0.08838835f;                 // 1/sqrt(128)
  #pragma unroll
  for (int t = 0; t < 8; t++) {
    v8f acc = {};
    #pragma unroll
    for (int s = 0; s < 4; s++) {
      AFrag b;
      load_bfrag(b, sT0, s, t, 8, lane);
      acc = __builtin_amdgcn_wmma_f32_16x16x32_bf16(false, aN[s].v, false, b.v,
                                                    (short)0, acc, false, false);
    }
    #pragma unroll
    for (int r = 0; r < 8; r++) {
      if (okReg[r])
        atomicAdd(&out[obReg[r] + 16*t + cl], acc[r] * inv);
    }
  }

  // --- Phase 4: f1 = (nf @ Wtp1) * Y1 * inv -> atomic scatter -------------
  #pragma unroll
  for (int t = 0; t < 4; t++) {
    v8f acc = {};
    #pragma unroll
    for (int s = 0; s < 4; s++) {
      AFrag b;
      load_bfrag(b, sT1, s, t, 4, lane);
      acc = __builtin_amdgcn_wmma_f32_16x16x32_bf16(false, aN[s].v, false, b.v,
                                                    (short)0, acc, false, false);
    }
    #pragma unroll
    for (int r = 0; r < 8; r++) {
      if (okReg[r]) {
        int   c = 16*t + cl;                     // channel in [0,64)
        float v = acc[r] * inv;
        float* base = &out[obReg[r] + 128 + 3*c];
        atomicAdd(base + 0, v * y1Reg[r][0]);
        atomicAdd(base + 1, v * y1Reg[r][1]);
        atomicAdd(base + 2, v * y1Reg[r][2]);
      }
    }
  }
}

// ---------------------------------------------------------------------------
// Kernel 2: per-node  out[:, :128] += B0 @ (Wmix0/sqrt(192))
//           B0 = [A0*A0, sum(A1*A1,-1)/sqrt(3)]  (192 wide)
// ---------------------------------------------------------------------------
__global__ __launch_bounds__(TPB)
void ace_node_kernel(const float* __restrict__ Wmix0,
                     float* __restrict__ out,
                     int N)
{
  __shared__ unsigned short sWm[192 * 128];       // 48 KB bf16, frag order
  __shared__ unsigned short sB0[WAVES][16 * 192]; // 48 KB per-wave staging

  const int tid  = threadIdx.x;
  const int wv   = tid >> 5;
  const int lane = tid & 31;
  const int hi   = lane >> 4;
  const int cl   = lane & 15;

  for (int idx = tid; idx < 192*128; idx += TPB) {
    int k = idx >> 7, n = idx & 127;
    sWm[bfrag_idx(k, n, 8)] = f32_to_bf16(Wmix0[idx]);
  }

  const int n0 = blockIdx.x * (WAVES * 16) + wv * 16;
  unsigned short* sb = sB0[wv];

  // Stage B0 (16 nodes x 192) as bf16.
  for (int m = 0; m < 16; m++) {
    int node = n0 + m;
    const float* row = out + (size_t)node * 320;
    for (int j = lane; j < 192; j += 32) {
      float bv = 0.f;
      if (node < N) {
        if (j < 128) {
          float v = row[j];
          bv = v * v;
        } else {
          const float* p = row + 128 + 3*(j - 128);
          bv = (p[0]*p[0] + p[1]*p[1] + p[2]*p[2]) * 0.57735027f; // 1/sqrt3
        }
      }
      sb[m*192 + j] = f32_to_bf16(bv);
    }
  }
  __syncthreads();

  AFrag aB[6];
  #pragma unroll
  for (int s = 0; s < 6; s++) load_afrag(aB[s], sb, cl, 192, 32*s, hi);

  const float scl = 0.07216878f;                  // 1/sqrt(192)
  #pragma unroll
  for (int t = 0; t < 8; t++) {
    v8f acc = {};
    #pragma unroll
    for (int s = 0; s < 6; s++) {
      AFrag b;
      load_bfrag(b, sWm, s, t, 8, lane);
      acc = __builtin_amdgcn_wmma_f32_16x16x32_bf16(false, aB[s].v, false, b.v,
                                                    (short)0, acc, false, false);
    }
    #pragma unroll
    for (int r = 0; r < 8; r++) {
      int node = n0 + r + 8*hi;
      if (node < N) {
        float* p = &out[(size_t)node*320 + 16*t + cl];
        *p += acc[r] * scl;                       // exclusive per (node,col)
      }
    }
  }
}

// ---------------------------------------------------------------------------
extern "C" void kernel_launch(void* const* d_in, const int* in_sizes, int n_in,
                              void* d_out, int out_size, void* d_ws, size_t ws_size,
                              hipStream_t stream) {
  const float* node_attrs = (const float*)d_in[0];
  const int*   edge_index = (const int*)  d_in[1];
  const float* edge_vec   = (const float*)d_in[2];
  const float* edge_len   = (const float*)d_in[3];
  const float* W1         = (const float*)d_in[4];
  const float* W2         = (const float*)d_in[5];
  const float* Wtp0       = (const float*)d_in[6];
  const float* Wtp1       = (const float*)d_in[7];
  const float* Wmix0      = (const float*)d_in[8];
  float* out = (float*)d_out;

  const int E = in_sizes[3];          // edge_len count
  const int N = in_sizes[0] / 128;    // node_attrs rows

  // Scatter target must start at zero (harness poisons d_out).
  hipMemsetAsync(d_out, 0, (size_t)out_size * sizeof(float), stream);

  const int eblocks = (E + WAVES*16 - 1) / (WAVES*16);
  ace_edge_kernel<<<eblocks, TPB, 0, stream>>>(
      node_attrs, edge_index, edge_index + E, edge_vec, edge_len,
      W1, W2, Wtp0, Wtp1, out, E);

  const int nblocks = (N + WAVES*16 - 1) / (WAVES*16);
  ace_node_kernel<<<nblocks, TPB, 0, stream>>>(Wmix0, out, N);
}
